// CausalStream_49400713838747
// MI455X (gfx1250) — matrix operand order repaired and checked by reading, not compiled, not run on hardware
//
#include <hip/hip_runtime.h>
#include <hip/hip_fp16.h>
#include <math.h>

// ---------------------------------------------------------------------------
// CausalStream on MI455X (gfx1250, wave32, WMMA)
// Sizes from reference: B=32, D=6, T=2048, FH=ENC=128, H=256, CONF=64, NC=10, K=5
// ---------------------------------------------------------------------------

#define BB   32
#define DD   6
#define TT   2048
#define FH   128
#define HH   256

typedef __attribute__((ext_vector_type(16))) _Float16 v16h;
typedef __attribute__((ext_vector_type(8)))  float    v8f;
typedef unsigned int u32x4 __attribute__((ext_vector_type(4)));
typedef int          i32x4 __attribute__((ext_vector_type(4)));
typedef int          i32x8 __attribute__((ext_vector_type(8)));

#define WMMA_F16(a, b, c) \
  __builtin_amdgcn_wmma_f32_16x16x32_f16(false, (a), false, (b), (short)0, (c), false, false)

#if defined(__has_builtin)
# if __has_builtin(__builtin_amdgcn_tensor_load_to_lds) && __has_builtin(__builtin_amdgcn_s_wait_tensorcnt)
#  define USE_TDM 1
# endif
#endif
#ifndef USE_TDM
# define USE_TDM 0
#endif

// ---------------------------------------------------------------------------
// Fragment helpers (layouts used in this file, wave32; A and B share one contiguous K order)
//   A 16x32 f16 : lane(lo,hi): M=lo; v16h[e]    = K 32c+16hi+e   (16 contiguous f16)
//   B 32x16 f16 : lane(lo,hi): N=lo; v16h[e]    = K 32c+16hi+e   (16 contiguous f16)
//   C/D 16x16 f32: v8f[r] -> M = r+8hi, N = lo
// ---------------------------------------------------------------------------
union AFrag { uint4 u4[2]; v16h h; };

// ===========================================================================
// Kernel 0a: pack conv weights (conv1..3) f32 (O,I,K) -> f16 [o][kc], kc=k*128+i
// ===========================================================================
__global__ __launch_bounds__(256) void pack_w_kernel(
    const float* __restrict__ w1, const float* __restrict__ w2,
    const float* __restrict__ w3, _Float16* __restrict__ wt) {
  int idx = blockIdx.x * 256 + threadIdx.x;             // 3*128*640 = 245760
  int l = idx / 81920, r = idx % 81920;
  int o = r / 640, kc = r % 640;
  int k = kc >> 7, i = kc & 127;
  const float* w = (l == 0) ? w1 : (l == 1) ? w2 : w3;
  wt[(size_t)l * 81920 + (size_t)o * 640 + kc] = (_Float16)w[(o * 128 + i) * 5 + k];
}

// ===========================================================================
// Kernel 0b: Wc = ip_w @ enc_w (f16, [h][c]) and fused bias
// ===========================================================================
__global__ __launch_bounds__(256) void wc_kernel(
    const float* __restrict__ ip_w, const float* __restrict__ enc_w,
    const float* __restrict__ enc_b, const float* __restrict__ ip_b,
    const float* __restrict__ state_bias,
    _Float16* __restrict__ Wc, float* __restrict__ biasc) {
  int idx = blockIdx.x * 256 + threadIdx.x;             // 256*128 = 32768
  int h = idx >> 7, c = idx & 127;
  float a = 0.f;
#pragma unroll 8
  for (int e = 0; e < 128; ++e) a += ip_w[h * 128 + e] * enc_w[e * 128 + c];
  Wc[h * 128 + c] = (_Float16)a;
  if (c == 0) {
    float bsum = ip_b[h] + state_bias[h];
#pragma unroll 8
    for (int e = 0; e < 128; ++e) bsum += ip_w[h * 128 + e] * enc_b[e];
    biasc[h] = bsum;
  }
}

// ===========================================================================
// Kernel 1: conv0 (D=6 -> FH=128, k=5, dil=1), plain VALU (tiny K), f16 out
// ===========================================================================
__global__ __launch_bounds__(256) void conv0_kernel(
    const float* __restrict__ M, const float* __restrict__ w0,
    const float* __restrict__ b0, _Float16* __restrict__ y0) {
  int b = blockIdx.y, t0 = blockIdx.x * 64;
  __shared__ float xs[6][68];                           // times t0-4 .. t0+63
  for (int idx = threadIdx.x; idx < 6 * 68; idx += 256) {
    int i = idx / 68, tau = idx % 68, t = t0 - 4 + tau;
    xs[i][tau] = (t >= 0) ? M[((size_t)b * 6 + i) * TT + t] : 0.f;
  }
  __syncthreads();
  int o = threadIdx.x & 127, half = threadIdx.x >> 7;
  float wr[30];
#pragma unroll
  for (int j = 0; j < 30; ++j) wr[j] = w0[o * 30 + j];
  float bias = b0[o];
#pragma unroll 1
  for (int s = 0; s < 32; ++s) {
    int lt = half * 32 + s;
    float acc = bias;
#pragma unroll
    for (int i = 0; i < 6; ++i)
#pragma unroll
      for (int k = 0; k < 5; ++k) acc += wr[i * 5 + k] * xs[i][lt + k];
    y0[((size_t)b * 128 + o) * TT + t0 + lt] = (_Float16)fmaxf(acc, 0.f);
  }
}

// ===========================================================================
// Kernel 2: dilated causal conv 128->128 via WMMA; K = 5 taps * 128 ch = 640
//   grid (T/16, B), 256 threads = 8 waves; wave w owns out-channels 16w..16w+15
// ===========================================================================
__global__ __launch_bounds__(256) void convN_kernel(
    const _Float16* __restrict__ src, _Float16* __restrict__ dst,
    const _Float16* __restrict__ wt, const float* __restrict__ bias, int dil) {
  int b = blockIdx.y, t0 = blockIdx.x * 16;
  int Tslab = 16 + 4 * dil;                             // <= 48
  __shared__ __align__(16) _Float16 xs[48 * 128];       // [tau][c], t-major
  for (int idx = threadIdx.x; idx < Tslab * 128; idx += 256) {
    int c = idx / Tslab, tau = idx % Tslab, t = t0 - 4 * dil + tau;
    xs[tau * 128 + c] = (t >= 0) ? src[((size_t)b * 128 + c) * TT + t] : (_Float16)0.f;
  }
  __syncthreads();
  int wave = threadIdx.x >> 5, lane = threadIdx.x & 31;
  int lo = lane & 15, hi = lane >> 4;
  int n = wave * 16 + lo;
  float bn = bias[n];
  v8f acc = {bn, bn, bn, bn, bn, bn, bn, bn};
#pragma unroll
  for (int c = 0; c < 20; ++c) {                        // 20 K-chunks of 32
    int k = c >> 2, ib = (c & 3) * 32;
    int row = (lo + k * dil) * 128;
    AFrag A;
    A.u4[0] = *(const uint4*)&xs[row + ib + hi * 16];
    A.u4[1] = *(const uint4*)&xs[row + ib + 8 + hi * 16];
    v16h Bf = *(const v16h*)&wt[(size_t)n * 640 + c * 32 + hi * 16];
    acc = WMMA_F16(A.h, Bf, acc);
  }
  union { uint4 u; _Float16 h8[8]; } st;
#pragma unroll
  for (int r = 0; r < 8; ++r) st.h8[r] = (_Float16)fmaxf(acc[r], 0.f);
  *(uint4*)(dst + ((size_t)b * 128 + n) * TT + t0 + hi * 8) = st.u;
}

// ===========================================================================
// Kernel 3: fused enc+ip projection  u[t][b][h] = Wc @ y3 + biasc  (f16 out)
//   grid (T/16, B), 512 threads = 16 waves (N=256)
// ===========================================================================
__global__ __launch_bounds__(512) void proj_kernel(
    const _Float16* __restrict__ y3, const _Float16* __restrict__ Wc,
    const float* __restrict__ biasc, _Float16* __restrict__ u) {
  int b = blockIdx.y, t0 = blockIdx.x * 16;
  __shared__ __align__(16) _Float16 ys[16 * 128];       // [tau][c]
  for (int idx = threadIdx.x; idx < 2048; idx += 512) {
    int c = idx >> 4, tau = idx & 15;
    ys[tau * 128 + c] = y3[((size_t)b * 128 + c) * TT + t0 + tau];
  }
  __syncthreads();
  int wave = threadIdx.x >> 5, lane = threadIdx.x & 31;
  int lo = lane & 15, hi = lane >> 4;
  int n = wave * 16 + lo;
  float bn = biasc[n];
  v8f acc = {bn, bn, bn, bn, bn, bn, bn, bn};
#pragma unroll
  for (int c = 0; c < 4; ++c) {
    AFrag A;
    A.u4[0] = *(const uint4*)&ys[lo * 128 + c * 32 + hi * 16];
    A.u4[1] = *(const uint4*)&ys[lo * 128 + c * 32 + 8 + hi * 16];
    v16h Bf = *(const v16h*)&Wc[(size_t)n * 128 + c * 32 + hi * 16];
    acc = WMMA_F16(A.h, Bf, acc);
  }
#pragma unroll
  for (int r = 0; r < 8; ++r)
    u[((size_t)(t0 + r + hi * 8) * BB + b) * HH + n] = (_Float16)acc[r];
}

// ===========================================================================
// Kernel 4: sequential SSM scan. One workgroup, 16 waves, T=2048 steps.
//   h_{t+1} = gelu(h_t @ A^T + u_t), hsum accumulated in registers.
//   u stream double-buffered into LDS via TDM (tensor_load_to_lds).
// ===========================================================================
#if USE_TDM
__device__ __forceinline__ void tdm_load_u_slab(const _Float16* gsrc, _Float16* ldst) {
  // 1-D tile: 32*256 f16 = 16384 B = 2048 x 8B elements (data_size=3)
  unsigned long long ga = (unsigned long long)(uintptr_t)gsrc;
  unsigned lds = (unsigned)(uintptr_t)ldst;             // low 32 bits = LDS byte addr
  u32x4 g0;
  g0[0] = 1u;                                           // count=1 (valid user D#)
  g0[1] = lds;                                          // lds_addr     [63:32]
  g0[2] = (unsigned)ga;                                 // global_addr  [95:64]
  g0[3] = (unsigned)((ga >> 32) & 0x01FFFFFFu) | (2u << 30); // [120:96] | type=2
  i32x8 g1;
  g1[0] = (int)(3u << 16);                              // data_size = 8B
  g1[1] = (int)(2048u << 16);                           // tensor_dim0 lo16 @ [63:48]
  g1[2] = (int)(1u << 16);                              // tensor_dim0 hi16 | tensor_dim1=1
  g1[3] = (int)(2048u << 16);                           // tile_dim0 = 2048
  g1[4] = 1;                                            // tile_dim1 = 1
  g1[5] = 2048;                                         // tensor_dim0_stride
  g1[6] = 0; g1[7] = 0;
  i32x4 z4 = {0, 0, 0, 0};
  i32x8 z8 = {0, 0, 0, 0, 0, 0, 0, 0};
  // 6-arg form (this toolchain): (g0, g1, g2, g3, g4, cpol)
  __builtin_amdgcn_tensor_load_to_lds(g0, g1, z4, z4, z8, 0);
}
#endif

__global__ __launch_bounds__(512) void scan_kernel(
    const float* __restrict__ Aw, const _Float16* __restrict__ u,
    float* __restrict__ hsum_out) {
  __shared__ __align__(16) _Float16 hbuf[2][32 * 256];  // h ping-pong (f16)
  __shared__ __align__(16) _Float16 ubuf[2][32 * 256];  // u_t double buffer
  int tid = threadIdx.x, wave = tid >> 5, lane = tid & 31;
  int lo = lane & 15, hi = lane >> 4;
  int n = wave * 16 + lo;                               // this wave's N column

  // Persistent B-fragments of A^T: bf[c][e] = A[n][32c + 16*hi + e]
  v16h bf[8];
#pragma unroll
  for (int c = 0; c < 8; ++c) {
    const float* ap = Aw + (size_t)n * HH + c * 32 + hi * 16;
#pragma unroll
    for (int e = 0; e < 16; ++e) bf[c][e] = (_Float16)ap[e];
  }

  for (int idx = tid; idx < 32 * 256; idx += 512) hbuf[0][idx] = (_Float16)0.f;
  v8f hs0 = {0.f, 0.f, 0.f, 0.f, 0.f, 0.f, 0.f, 0.f};
  v8f hs1 = hs0;

#if USE_TDM
  if (wave == 0) {
    tdm_load_u_slab(u, &ubuf[0][0]);
    __builtin_amdgcn_s_wait_tensorcnt(0);
  }
#else
  for (int idx = tid; idx < 8192; idx += 512) ubuf[0][idx] = u[idx];
#endif
  __syncthreads();

#pragma unroll 1
  for (int t = 0; t < TT; ++t) {
    const _Float16* uc = ubuf[t & 1];
    const _Float16* hc = hbuf[t & 1];
    _Float16* hn = hbuf[(t + 1) & 1];
#if USE_TDM
    if (wave == 0 && t + 1 < TT)
      tdm_load_u_slab(u + (size_t)(t + 1) * 8192, &ubuf[(t + 1) & 1][0]);
#else
    if (t + 1 < TT)
      for (int idx = tid; idx < 8192; idx += 512)
        ubuf[(t + 1) & 1][idx] = u[(size_t)(t + 1) * 8192 + idx];
#endif
#pragma unroll
    for (int m = 0; m < 2; ++m) {                       // two 16-row b-tiles
      v8f acc;
#pragma unroll
      for (int r = 0; r < 8; ++r)
        acc[r] = (float)uc[(m * 16 + hi * 8 + r) * 256 + n];
#pragma unroll
      for (int c = 0; c < 8; ++c) {
        AFrag A;
        A.u4[0] = *(const uint4*)&hc[(m * 16 + lo) * 256 + c * 32 + hi * 16];
        A.u4[1] = *(const uint4*)&hc[(m * 16 + lo) * 256 + c * 32 + 8 + hi * 16];
        acc = WMMA_F16(A.h, bf[c], acc);
      }
#pragma unroll
      for (int r = 0; r < 8; ++r) {                     // exact GELU
        float x = acc[r];
        acc[r] = 0.5f * x * (1.f + erff(x * 0.70710678118654752f));
      }
      if (m == 0) hs0 += acc; else hs1 += acc;
#pragma unroll
      for (int r = 0; r < 8; ++r)
        hn[(m * 16 + hi * 8 + r) * 256 + n] = (_Float16)acc[r];
    }
#if USE_TDM
    if (wave == 0) __builtin_amdgcn_s_wait_tensorcnt(0);
#endif
    __syncthreads();
  }
#pragma unroll
  for (int r = 0; r < 8; ++r) {
    hsum_out[(hi * 8 + r) * HH + n] = hs0[r];
    hsum_out[(16 + hi * 8 + r) * HH + n] = hs1[r];
  }
}

// ===========================================================================
// Kernel 5: per-(b,d) mean + unbiased std of M over T  ->  s[b][0..11]
// ===========================================================================
__global__ __launch_bounds__(256) void stats_kernel(
    const float* __restrict__ M, float* __restrict__ sout) {
  int b = blockIdx.x, tid = threadIdx.x;
  __shared__ float r1[256], r2[256];
#pragma unroll 1
  for (int d = 0; d < 6; ++d) {
    float s1 = 0.f, s2 = 0.f;
    const float* row = M + ((size_t)b * 6 + d) * TT;
    for (int t = tid; t < TT; t += 256) { float v = row[t]; s1 += v; s2 += v * v; }
    r1[tid] = s1; r2[tid] = s2;
    __syncthreads();
    for (int o = 128; o > 0; o >>= 1) {
      if (tid < o) { r1[tid] += r1[tid + o]; r2[tid] += r2[tid + o]; }
      __syncthreads();
    }
    if (tid == 0) {
      float mean = r1[0] * (1.f / TT);
      float var = (r2[0] - (float)TT * mean * mean) * (1.f / (TT - 1));
      sout[b * 12 + d] = mean;
      sout[b * 12 + 6 + d] = sqrtf(fmaxf(var, 0.f));
    }
    __syncthreads();
  }
}

// ===========================================================================
// Kernel 6: tail — pooled, classifier, metanet, frontdoor, corrected.
//   outputs flat: logits[320] | pooled[8192] | corrected[8192] | conf[2048]
// ===========================================================================
__global__ __launch_bounds__(256) void tail_kernel(
    const float* __restrict__ hsum, const float* __restrict__ sst,
    const float* __restrict__ op_w, const float* __restrict__ op_b,
    const float* __restrict__ cls_w1, const float* __restrict__ cls_b1,
    const float* __restrict__ cls_w2, const float* __restrict__ cls_b2,
    const float* __restrict__ mn_w1, const float* __restrict__ mn_b1,
    const float* __restrict__ mn_w2, const float* __restrict__ mn_b2,
    const float* __restrict__ mn_w3, const float* __restrict__ mn_b3,
    const float* __restrict__ fd_w1, const float* __restrict__ fd_b1,
    const float* __restrict__ fd_w2, const float* __restrict__ fd_b2,
    float* __restrict__ out) {
  __shared__ float pooled_s[8192];
  __shared__ float tmp[8192];
  int tid = threadIdx.x;
  const float invT = 1.f / (float)TT;
  // pooled = (hsum/T) @ op_w^T + op_b
  for (int idx = tid; idx < 8192; idx += 256) {
    int b = idx >> 8, h = idx & 255;
    float a = 0.f;
#pragma unroll 8
    for (int k = 0; k < 256; ++k) a += hsum[b * 256 + k] * op_w[h * 256 + k];
    a = a * invT + op_b[h];
    pooled_s[idx] = a;
    out[320 + idx] = a;
  }
  __syncthreads();
  // hid = relu(pooled @ cls_w1^T + b1)
  for (int idx = tid; idx < 8192; idx += 256) {
    int b = idx >> 8, h = idx & 255;
    float a = cls_b1[h];
#pragma unroll 8
    for (int k = 0; k < 256; ++k) a += pooled_s[b * 256 + k] * cls_w1[h * 256 + k];
    tmp[idx] = fmaxf(a, 0.f);
  }
  __syncthreads();
  // logits
  for (int idx = tid; idx < 320; idx += 256) {
    int b = idx / 10, j = idx % 10;
    float a = cls_b2[j];
#pragma unroll 8
    for (int k = 0; k < 256; ++k) a += tmp[b * 256 + k] * cls_w2[j * 256 + k];
    out[idx] = a;
  }
  __syncthreads();
  // metanet z1 = relu(s @ mn_w1^T + b1): tmp[0:2048)
  for (int idx = tid; idx < 2048; idx += 256) {
    int b = idx >> 6, j = idx & 63;
    float a = mn_b1[j];
#pragma unroll
    for (int k = 0; k < 12; ++k) a += sst[b * 12 + k] * mn_w1[j * 12 + k];
    tmp[idx] = fmaxf(a, 0.f);
  }
  __syncthreads();
  // z2 = relu(z1 @ mn_w2^T + b2): tmp[2048:6144)
  for (int idx = tid; idx < 4096; idx += 256) {
    int b = idx >> 7, j = idx & 127;
    float a = mn_b2[j];
#pragma unroll 8
    for (int k = 0; k < 64; ++k) a += tmp[b * 64 + k] * mn_w2[j * 64 + k];
    tmp[2048 + idx] = fmaxf(a, 0.f);
  }
  __syncthreads();
  // conf_hat = z2 @ mn_w3^T + b3: tmp[6144:8192), also output
  for (int idx = tid; idx < 2048; idx += 256) {
    int b = idx >> 6, j = idx & 63;
    float a = mn_b3[j];
#pragma unroll 8
    for (int k = 0; k < 128; ++k) a += tmp[2048 + b * 128 + k] * mn_w3[j * 128 + k];
    tmp[6144 + idx] = a;
    out[16704 + idx] = a;
  }
  __syncthreads();
  // fd1 = relu(conf @ fd_w1^T + b1): tmp[0:4096)
  for (int idx = tid; idx < 4096; idx += 256) {
    int b = idx >> 7, j = idx & 127;
    float a = fd_b1[j];
#pragma unroll 8
    for (int k = 0; k < 64; ++k) a += tmp[6144 + b * 64 + k] * fd_w1[j * 64 + k];
    tmp[idx] = fmaxf(a, 0.f);
  }
  __syncthreads();
  // corrected = pooled - (fd1 @ fd_w2^T + b2)
  for (int idx = tid; idx < 8192; idx += 256) {
    int b = idx >> 8, h = idx & 255;
    float a = fd_b2[h];
#pragma unroll 8
    for (int k = 0; k < 128; ++k) a += tmp[b * 128 + k] * fd_w2[h * 128 + k];
    out[8512 + idx] = pooled_s[idx] - a;
  }
}

// ===========================================================================
// Host launch
// ===========================================================================
extern "C" void kernel_launch(void* const* d_in, const int* in_sizes, int n_in,
                              void* d_out, int out_size, void* d_ws, size_t ws_size,
                              hipStream_t stream) {
  (void)in_sizes; (void)n_in; (void)out_size; (void)ws_size;
  const float* M      = (const float*)d_in[0];
  const float* cw0    = (const float*)d_in[1];  const float* cb0 = (const float*)d_in[2];
  const float* cw1    = (const float*)d_in[3];  const float* cb1 = (const float*)d_in[4];
  const float* cw2    = (const float*)d_in[5];  const float* cb2 = (const float*)d_in[6];
  const float* cw3    = (const float*)d_in[7];  const float* cb3 = (const float*)d_in[8];
  const float* enc_w  = (const float*)d_in[9];  const float* enc_b = (const float*)d_in[10];
  const float* Aw     = (const float*)d_in[11];
  const float* ip_w   = (const float*)d_in[12]; const float* ip_b  = (const float*)d_in[13];
  const float* sbias  = (const float*)d_in[14];
  const float* op_w   = (const float*)d_in[15]; const float* op_b  = (const float*)d_in[16];
  const float* cls_w1 = (const float*)d_in[17]; const float* cls_b1 = (const float*)d_in[18];
  const float* cls_w2 = (const float*)d_in[19]; const float* cls_b2 = (const float*)d_in[20];
  const float* mn_w1  = (const float*)d_in[21]; const float* mn_b1 = (const float*)d_in[22];
  const float* mn_w2  = (const float*)d_in[23]; const float* mn_b2 = (const float*)d_in[24];
  const float* mn_w3  = (const float*)d_in[25]; const float* mn_b3 = (const float*)d_in[26];
  const float* fd_w1  = (const float*)d_in[27]; const float* fd_b1 = (const float*)d_in[28];
  const float* fd_w2  = (const float*)d_in[29]; const float* fd_b2 = (const float*)d_in[30];
  float* out = (float*)d_out;

  // Workspace layout (bytes); total ~67.7 MB
  char* w = (char*)d_ws;
  _Float16* yA    = (_Float16*)(w + 0);            // 32*128*2048 f16 = 16 MB
  _Float16* yB    = (_Float16*)(w + (16u << 20));  // 16 MB
  _Float16* u     = (_Float16*)(w + (32u << 20));  // (T,B,H) f16 = 32 MB
  _Float16* wt    = (_Float16*)(w + (64u << 20));  // 3*128*640 f16
  _Float16* Wc    = (_Float16*)(w + (64u << 20) + 524288);
  float*    biasc = (float*)   (w + (64u << 20) + 524288 + 65536);
  float*    hsum  = (float*)   (w + (64u << 20) + 524288 + 65536 + 4096);
  float*    sst   = (float*)   (w + (64u << 20) + 524288 + 65536 + 4096 + 32768);

  pack_w_kernel<<<960, 256, 0, stream>>>(cw1, cw2, cw3, wt);
  wc_kernel<<<128, 256, 0, stream>>>(ip_w, enc_w, enc_b, ip_b, sbias, Wc, biasc);
  conv0_kernel<<<dim3(TT / 64, BB), 256, 0, stream>>>(M, cw0, cb0, yA);
  convN_kernel<<<dim3(TT / 16, BB), 256, 0, stream>>>(yA, yB, wt,             cb1, 2);
  convN_kernel<<<dim3(TT / 16, BB), 256, 0, stream>>>(yB, yA, wt + 81920,     cb2, 4);
  convN_kernel<<<dim3(TT / 16, BB), 256, 0, stream>>>(yA, yB, wt + 2 * 81920, cb3, 8);
  proj_kernel<<<dim3(TT / 16, BB), 512, 0, stream>>>(yB, Wc, biasc, u);
  scan_kernel<<<1, 512, 0, stream>>>(Aw, u, hsum);
  stats_kernel<<<BB, 256, 0, stream>>>(M, sst);
  tail_kernel<<<1, 256, 0, stream>>>(hsum, sst, op_w, op_b, cls_w1, cls_b1,
                                     cls_w2, cls_b2, mn_w1, mn_b1, mn_w2, mn_b2,
                                     mn_w3, mn_b3, fd_w1, fd_b1, fd_w2, fd_b2, out);
}
